// PredCells_19524921327865
// MI455X (gfx1250) — matrix operand order repaired and checked by reading, not compiled, not run on hardware
//
#include <hip/hip_runtime.h>
#include <stdint.h>

// ---------------- problem constants ----------------
#define H    1024
#define G4   4096          // 4*H
#define CC   256           // numchars
#define TT   512           // timesteps
#define LAMF 0.05f

// ---------------- launch constants -----------------
#define NB   32            // persistent blocks (must be co-resident)
#define NT   256           // threads per block (8 wave32)
#define NWAVES ((NB * NT) / 32)

// ---------------- CDNA5 WMMA types -----------------
typedef __bf16 bf16;
typedef bf16  v16bf __attribute__((ext_vector_type(16)));
typedef float v8f   __attribute__((ext_vector_type(8)));
typedef int   v4i   __attribute__((ext_vector_type(4)));
typedef unsigned short u16;

// =====================================================================
// helpers
// =====================================================================
__device__ __forceinline__ float sigm(float x) { return 1.f / (1.f + expf(-x)); }

__device__ __forceinline__ float wred32(float v) {
#pragma unroll
  for (int o = 16; o > 0; o >>= 1) v += __shfl_down(v, o, 32);
  return v;
}

__device__ __forceinline__ uint32_t f2bf_bits(float x) {  // RNE fp32->bf16
  union { float f; uint32_t u; } c; c.f = x;
  uint32_t r = c.u + 0x7fffu + ((c.u >> 16) & 1u);
  return r >> 16;
}

// device-scope sense barrier for the persistent kernel
__device__ __forceinline__ void gsync(unsigned int* cnt, unsigned int* gen) {
  __syncthreads();
  if (threadIdx.x == 0) {
    unsigned int g = __hip_atomic_load(gen, __ATOMIC_ACQUIRE, __HIP_MEMORY_SCOPE_AGENT);
    unsigned int a = __hip_atomic_fetch_add(cnt, 1u, __ATOMIC_ACQ_REL, __HIP_MEMORY_SCOPE_AGENT);
    if (a == (unsigned int)(NB - 1)) {
      __hip_atomic_store(cnt, 0u, __ATOMIC_RELAXED, __HIP_MEMORY_SCOPE_AGENT);
      __hip_atomic_fetch_add(gen, 1u, __ATOMIC_ACQ_REL, __HIP_MEMORY_SCOPE_AGENT);
    } else {
      while (__hip_atomic_load(gen, __ATOMIC_ACQUIRE, __HIP_MEMORY_SCOPE_AGENT) == g)
        __builtin_amdgcn_s_sleep(2);
    }
  }
  __syncthreads();
}

// =====================================================================
// WMMA GEMV:  out[M] += Wbf16[M x K] * z[K]   (out pre-initialized w/ bias)
//  - A fragments pre-swizzled to the exact CDNA5 16-bit A layout; each
//    lane reads 8 consecutive dwords (two b128 loads) per 16x32 tile.
//  - zb is the activation vector ALREADY packed bf16: B column 0 filled
//    by lanes 0 (K=0..7,16..23) and 16 (K=8..15,24..31) with two b128 loads, no cvt.
//  - If bf_out != nullptr the writer lanes also emit a packed-bf16 copy
//    of the result (used when the output feeds a later GEMV).
// =====================================================================
__device__ __forceinline__ void wmma_matvec(const uint32_t* __restrict__ frag,
                                            int Mtiles, int Ktiles,
                                            const u16* __restrict__ zb,
                                            float* __restrict__ out,
                                            u16* __restrict__ bf_out,
                                            int wave, int nWaves, int lane) {
  const bool writer = ((lane & 15) == 0);
  for (int mt = wave; mt < Mtiles; mt += nWaves) {
    v8f acc;
#pragma unroll
    for (int r = 0; r < 8; ++r) acc[r] = 0.f;
    v16bf b;
#pragma unroll
    for (int e = 0; e < 16; ++e) b[e] = (bf16)0.f;   // non-writer lanes stay 0
    const uint32_t* tp = frag + (size_t)mt * Ktiles * 256 + lane * 8;
    for (int kt = 0; kt < Ktiles; ++kt) {
      v16bf a;
      {
        v4i t0 = *(const v4i*)(tp);
        v4i t1 = *(const v4i*)(tp + 4);
        v4i* ap = (v4i*)&a; ap[0] = t0; ap[1] = t1;
      }
      tp += 256;
      __builtin_prefetch((const void*)(tp + 256), 0, 3);  // WGP-scope prefetch
      if (writer) {
        const u16* zz = zb + kt * 32 + ((lane & 16) >> 1);
        v4i* bp = (v4i*)&b;
        bp[0] = *(const v4i*)(zz);
        bp[1] = *(const v4i*)(zz + 16);
      }
      acc = __builtin_amdgcn_wmma_f32_16x16x32_bf16(false, a, false, b,
                                                    (short)0, acc, false, false);
    }
    if (writer) {  // lane0 -> rows 0..7, lane16 -> rows 8..15
      int rb = mt * 16 + ((lane & 16) ? 8 : 0);
      float* op = out + rb;
#pragma unroll
      for (int r = 0; r < 8; ++r) {
        float v = op[r] + acc[r];
        op[r] = v;
        if (bf_out) bf_out[rb + r] = (u16)f2bf_bits(v);
      }
    }
  }
}

// =====================================================================
// pack fp32 [M x K] row-major -> bf16 A-fragment stream
// =====================================================================
__global__ void pack_bf16_frags(const float* __restrict__ src, int M, int K,
                                uint32_t* __restrict__ dst, int ktOff, int ktTot) {
  int ktLoc = K >> 5;
  size_t total = (size_t)(M >> 4) * ktLoc * 256;
  for (size_t gid = (size_t)blockIdx.x * blockDim.x + threadIdx.x; gid < total;
       gid += (size_t)gridDim.x * blockDim.x) {
    int within = (int)(gid & 255);
    size_t tile = gid >> 8;
    int lane = within >> 3;
    int v    = within & 7;
    int ktl  = (int)(tile % ktLoc);
    int mtt  = (int)(tile / ktLoc);
    int row  = mtt * 16 + (lane & 15);
    int hi   = (lane >> 4) & 1;
    int kb   = (v < 4) ? (hi * 8 + 2 * v) : (16 + hi * 8 + 2 * (v - 4));
    int k    = ktl * 32 + kb;
    uint32_t lo = f2bf_bits(src[(size_t)row * K + k]);
    uint32_t hi16 = f2bf_bits(src[(size_t)row * K + k + 1]);
    dst[(((size_t)mtt * ktTot + (ktOff + ktl)) * 32 + lane) * 8 + v] = lo | (hi16 << 16);
  }
}

__global__ void add_vec(const float* __restrict__ a, const float* __restrict__ b,
                        float* __restrict__ o, int n) {
  int t = blockIdx.x * blockDim.x + threadIdx.x;
  if (t < n) o[t] = a[t] + b[t];
}

__global__ void init_state(float* z1, float* z2, float* z3,
                           u16* z1b, u16* z2b, u16* z3b,
                           float* c1, float* c2, float* c3,
                           float* rec2, float* rec3,
                           const float* r2, const float* r3, const float* Wb0,
                           unsigned int* cnt, unsigned int* gen,
                           float* lossAcc, float* fllAcc) {
  int t = blockIdx.x * blockDim.x + threadIdx.x;
  if (t < 3 * H) {
    z1[t] = (t < H) ? Wb0[t] : 0.f;
    z2[t] = 0.f;
    z1b[t] = 0; z2b[t] = 0;
  }
  if (t < 2 * H) { z3[t] = 0.f; z3b[t] = 0; }
  if (t < H) { c1[t] = 0.f; c2[t] = 0.f; c3[t] = 0.f; rec2[t] = r2[t]; rec3[t] = r3[t]; }
  if (t == 0) { *cnt = 0u; *gen = 0u; *lossAcc = 0.f; *fllAcc = 0.f; }
}

// =====================================================================
// persistent kernel
// =====================================================================
struct Ptrs {
  // inputs (fp32)
  const float *x, *Vw1, *Vb1, *Vb2, *Vb3, *Ww0, *Wb0, *Wb1, *Wb2, *r1;
  // packed bf16 weight fragments
  const uint32_t *G1, *G2, *G3, *W1, *W2, *V2, *V3;
  // fp32 state / scratch
  float *z1, *z2, *z3, *g, *c1, *c2, *c3, *rec1, *rec2, *rec3, *td0, *b1, *b2, *b3;
  // packed bf16 activation shadows
  u16 *z1b, *z2b, *z3b;
  unsigned int *cnt, *gen;
  float *lossAcc, *fllAcc;
  float *out;
};

__global__ __launch_bounds__(NT, 1) void predcells_persistent(Ptrs p) {
  __shared__ float sbuf[CC];
  const int tid  = blockIdx.x * NT + threadIdx.x;
  const int lane = threadIdx.x & 31;
  const int wave = tid >> 5;

  // ---- init: block0 computes rec1 = softmax(r1), td0, loss0(t=0) ----
  if (blockIdx.x == 0) {
    int r = threadIdx.x;  // 0..255
    float v = p.r1[r];
    sbuf[r] = v; __syncthreads();
    for (int s = 128; s > 0; s >>= 1) { if (r < s) sbuf[r] = fmaxf(sbuf[r], sbuf[r + s]); __syncthreads(); }
    float mx = sbuf[0]; __syncthreads();
    float e = expf(v - mx); sbuf[r] = e; __syncthreads();
    for (int s = 128; s > 0; s >>= 1) { if (r < s) sbuf[r] += sbuf[r + s]; __syncthreads(); }
    float rr = e / sbuf[0]; __syncthreads();
    p.rec1[r] = rr;
    float s0 = p.x[r];
    p.td0[r] = s0 - rr;
    float u  = s0 * logf(rr) + (1.f - s0) * logf(1.f - rr);
    float red = wred32(-u);
    if (lane == 0) { atomicAdd(p.lossAcc, red); atomicAdd(p.fllAcc, red); }
  }
  gsync(p.cnt, p.gen);

  for (int t = 0; t < TT; ++t) {
    // ---- P0: rec3n (t>0) | BU0 | g <- bias1 ----
    if (wave < 64) {
      if (t > 0) wmma_matvec(p.V3, 64, 32, p.z3b + H, p.rec3, nullptr, wave, 64, lane);
    } else if (wave < 96) {
      int row = (wave - 64) * 32 + lane;
      const float* wr = p.Ww0 + (size_t)row * CC;
      float acc = 0.f;
      for (int k = 0; k < CC; k += 4)
        acc += wr[k] * p.td0[k] + wr[k + 1] * p.td0[k + 1] +
               wr[k + 2] * p.td0[k + 2] + wr[k + 3] * p.td0[k + 3];
      float v = p.z1[row] + acc;          // z1[0:H] pre-set to Wb0
      p.z1[row] = v;
      p.z1b[row] = (u16)f2bf_bits(v);
    } else if (wave < 128) {
      int idx = (wave - 96) * 32 + lane;
      for (int j = idx; j < G4; j += 1024) p.g[j] = p.b1[j];
    }
    gsync(p.cnt, p.gen);

    // ---- P1: gates L1 = [W_ih_1|W_hh_1] @ [BU0;TD1;h1] ----
    wmma_matvec(p.G1, 256, 96, p.z1b, p.g, nullptr, wave, NWAVES, lane);
    gsync(p.cnt, p.gen);

    // ---- P2: LSTM1 pointwise, TD1n, pre-inits ----
    if (tid < H) {
      int j = tid;
      float gi = p.g[j], gf = p.g[j + H], gg = p.g[j + 2 * H], go = p.g[j + 3 * H];
      float c = sigm(gf) * p.c1[j] + sigm(gi) * tanhf(gg);
      p.c1[j] = c;
      float h = sigm(go) * tanhf(c);
      p.z1[2 * H + j] = h;                 // fp32 h1 (block0 logits use it)
      p.z1b[2 * H + j] = (u16)f2bf_bits(h);
      float td = h - p.rec2[j];
      p.z1b[H + j] = (u16)f2bf_bits(td);
      p.rec2[j] = p.Vb2[j];                // target of rec2n accumulation
      p.z2[j]   = p.Wb1[j];                // target of BU1 accumulation
      p.g[j] = p.b2[j]; p.g[j + H] = p.b2[j + H];
      p.g[j + 2 * H] = p.b2[j + 2 * H]; p.g[j + 3 * H] = p.b2[j + 3 * H];
      float red = wred32(fabsf(td));
      if (lane == 0) atomicAdd(p.lossAcc, LAMF * red);
    }
    gsync(p.cnt, p.gen);

    // ---- P3: block0: logits+softmax+prediction ; others: BU1 = Ww_1 @ TD1n ----
    if (blockIdx.x == 0) {
      int r = threadIdx.x;
      const float* vr = p.Vw1 + (size_t)r * H;
      const float* hh = p.z1 + 2 * H;
      float v = p.Vb1[r];
      for (int k = 0; k < H; k += 4)
        v += vr[k] * hh[k] + vr[k + 1] * hh[k + 1] + vr[k + 2] * hh[k + 2] + vr[k + 3] * hh[k + 3];
      sbuf[r] = v; __syncthreads();
      for (int s = 128; s > 0; s >>= 1) { if (r < s) sbuf[r] = fmaxf(sbuf[r], sbuf[r + s]); __syncthreads(); }
      float mx = sbuf[0]; __syncthreads();
      float e = expf(v - mx); sbuf[r] = e; __syncthreads();
      for (int s = 128; s > 0; s >>= 1) { if (r < s) sbuf[r] += sbuf[r + s]; __syncthreads(); }
      float rr = e / sbuf[0]; __syncthreads();
      p.rec1[r] = rr;
      p.out[2 + (size_t)t * CC + r] = rr;  // prediction at t
    } else {
      wmma_matvec(p.W1, 64, 32, p.z1b + H, p.z2, p.z2b,
                  wave - (NT / 32), NWAVES - (NT / 32), lane);
    }
    gsync(p.cnt, p.gen);

    // ---- P4: gates L2 = [W_ih_2|W_hh_2] @ [BU1;TD2;h2] ----
    wmma_matvec(p.G2, 256, 96, p.z2b, p.g, nullptr, wave, NWAVES, lane);
    gsync(p.cnt, p.gen);

    // ---- P5: LSTM2 pointwise, TD2n, pre-inits ----
    if (tid < H) {
      int j = tid;
      float gi = p.g[j], gf = p.g[j + H], gg = p.g[j + 2 * H], go = p.g[j + 3 * H];
      float c = sigm(gf) * p.c2[j] + sigm(gi) * tanhf(gg);
      p.c2[j] = c;
      float h = sigm(go) * tanhf(c);
      p.z2b[2 * H + j] = (u16)f2bf_bits(h);
      float td = h - p.rec3[j];
      p.z2b[H + j] = (u16)f2bf_bits(td);
      p.rec3[j] = p.Vb3[j];                // target of rec3n accumulation
      p.z3[j]   = p.Wb2[j];                // target of BU2 accumulation
      p.g[j] = p.b3[j]; p.g[j + H] = p.b3[j + H];
      p.g[j + 2 * H] = p.b3[j + 2 * H]; p.g[j + 3 * H] = p.b3[j + 3 * H];
      float red = wred32(fabsf(td));
      if (lane == 0) atomicAdd(p.lossAcc, LAMF * LAMF * red);
    }
    gsync(p.cnt, p.gen);

    // ---- P6: BU2 = Ww_2 @ TD2n ; rec2n = Vw_2 @ h2n ----
    {
      int hw = NWAVES / 2;
      if (wave < hw)
        wmma_matvec(p.W2, 64, 32, p.z2b + H,     p.z3,   p.z3b,  wave,      hw, lane);
      else
        wmma_matvec(p.V2, 64, 32, p.z2b + 2 * H, p.rec2, nullptr, wave - hw, hw, lane);
    }
    gsync(p.cnt, p.gen);

    // ---- P7: gates L3 = [W_ih_3|W_hh_3] @ [BU2;h3] ----
    wmma_matvec(p.G3, 256, 64, p.z3b, p.g, nullptr, wave, NWAVES, lane);
    gsync(p.cnt, p.gen);

    // ---- P8: LSTM3 pointwise + next-step prep (td0, loss0, z1<-Wb0) ----
    if (tid < H) {
      int j = tid;
      float gi = p.g[j], gf = p.g[j + H], gg = p.g[j + 2 * H], go = p.g[j + 3 * H];
      float c = sigm(gf) * p.c3[j] + sigm(gi) * tanhf(gg);
      p.c3[j] = c;
      float h = sigm(go) * tanhf(c);
      p.z3b[H + j] = (u16)f2bf_bits(h);
      if (t + 1 < TT) p.z1[j] = p.Wb0[j];
    }
    if (t + 1 < TT && tid < CC) {
      float rc = p.rec1[tid];
      float s0 = p.x[(size_t)(t + 1) * CC + tid];
      p.td0[tid] = s0 - rc;
      float u = s0 * logf(rc) + (1.f - s0) * logf(1.f - rc);
      float red = wred32(-u);
      if (lane == 0) { atomicAdd(p.lossAcc, red); atomicAdd(p.fllAcc, red); }
    }
    gsync(p.cnt, p.gen);
  }

  if (tid == 0) { p.out[0] = *p.lossAcc; p.out[1] = *p.fllAcc; }
}

// =====================================================================
// host launcher
// =====================================================================
extern "C" void kernel_launch(void* const* d_in, const int* in_sizes, int n_in,
                              void* d_out, int out_size, void* d_ws, size_t ws_size,
                              hipStream_t stream) {
  (void)in_sizes; (void)n_in; (void)out_size; (void)ws_size;
  const float* x      = (const float*)d_in[0];
  const float* Wih1   = (const float*)d_in[1];
  const float* Whh1   = (const float*)d_in[2];
  const float* bih1   = (const float*)d_in[3];
  const float* bhh1   = (const float*)d_in[4];
  const float* Vw1    = (const float*)d_in[5];
  const float* Vb1    = (const float*)d_in[6];
  const float* Wih2   = (const float*)d_in[7];
  const float* Whh2   = (const float*)d_in[8];
  const float* bih2   = (const float*)d_in[9];
  const float* bhh2   = (const float*)d_in[10];
  const float* Vw2    = (const float*)d_in[11];
  const float* Vb2    = (const float*)d_in[12];
  const float* Wih3   = (const float*)d_in[13];
  const float* Whh3   = (const float*)d_in[14];
  const float* bih3   = (const float*)d_in[15];
  const float* bhh3   = (const float*)d_in[16];
  const float* Vw3    = (const float*)d_in[17];
  const float* Vb3    = (const float*)d_in[18];
  const float* Ww0    = (const float*)d_in[19];
  const float* Wb0    = (const float*)d_in[20];
  const float* Ww1    = (const float*)d_in[21];
  const float* Wb1    = (const float*)d_in[22];
  const float* Ww2    = (const float*)d_in[23];
  const float* Wb2    = (const float*)d_in[24];
  const float* r1     = (const float*)d_in[25];
  const float* r2     = (const float*)d_in[26];
  const float* r3     = (const float*)d_in[27];

  uint8_t* ws = (uint8_t*)d_ws;
  size_t off = 0;
  auto A = [&](size_t bytes) -> void* {
    void* p = ws + off;
    off += (bytes + 255) & ~(size_t)255;
    return p;
  };
  uint32_t* G1p = (uint32_t*)A((size_t)G4 * 3072 * 2);   // 24 MB
  uint32_t* G2p = (uint32_t*)A((size_t)G4 * 3072 * 2);   // 24 MB
  uint32_t* G3p = (uint32_t*)A((size_t)G4 * 2048 * 2);   // 16 MB
  uint32_t* W1p = (uint32_t*)A((size_t)H * H * 2);
  uint32_t* W2p = (uint32_t*)A((size_t)H * H * 2);
  uint32_t* V2p = (uint32_t*)A((size_t)H * H * 2);
  uint32_t* V3p = (uint32_t*)A((size_t)H * H * 2);
  float* z1  = (float*)A(3 * H * 4);
  float* z2  = (float*)A(3 * H * 4);
  float* z3  = (float*)A(2 * H * 4);
  u16*   z1b = (u16*)A(3 * H * 2);
  u16*   z2b = (u16*)A(3 * H * 2);
  u16*   z3b = (u16*)A(2 * H * 2);
  float* g   = (float*)A(G4 * 4);
  float* c1  = (float*)A(H * 4);
  float* c2  = (float*)A(H * 4);
  float* c3  = (float*)A(H * 4);
  float* rec1 = (float*)A(CC * 4);
  float* rec2 = (float*)A(H * 4);
  float* rec3 = (float*)A(H * 4);
  float* td0  = (float*)A(CC * 4);
  float* b1   = (float*)A(G4 * 4);
  float* b2   = (float*)A(G4 * 4);
  float* b3   = (float*)A(G4 * 4);
  uint8_t* ctrl = (uint8_t*)A(256);
  unsigned int* cnt = (unsigned int*)(ctrl);
  unsigned int* gen = (unsigned int*)(ctrl + 64);
  float* lossAcc    = (float*)(ctrl + 128);
  float* fllAcc     = (float*)(ctrl + 132);

  auto pack = [&](const float* src, int M, int K, uint32_t* dst, int ktOff, int ktTot) {
    size_t total = (size_t)(M / 16) * (K / 32) * 256;
    int blocks = (int)((total + 255) / 256);
    pack_bf16_frags<<<blocks, 256, 0, stream>>>(src, M, K, dst, ktOff, ktTot);
  };
  pack(Wih1, G4, 2048, G1p, 0, 96);  pack(Whh1, G4, 1024, G1p, 64, 96);
  pack(Wih2, G4, 2048, G2p, 0, 96);  pack(Whh2, G4, 1024, G2p, 64, 96);
  pack(Wih3, G4, 1024, G3p, 0, 64);  pack(Whh3, G4, 1024, G3p, 32, 64);
  pack(Ww1,  H,  1024, W1p, 0, 32);  pack(Ww2,  H,  1024, W2p, 0, 32);
  pack(Vw2,  H,  1024, V2p, 0, 32);  pack(Vw3,  H,  1024, V3p, 0, 32);

  add_vec<<<(G4 + 255) / 256, 256, 0, stream>>>(bih1, bhh1, b1, G4);
  add_vec<<<(G4 + 255) / 256, 256, 0, stream>>>(bih2, bhh2, b2, G4);
  add_vec<<<(G4 + 255) / 256, 256, 0, stream>>>(bih3, bhh3, b3, G4);

  init_state<<<(3 * H + 255) / 256, 256, 0, stream>>>(z1, z2, z3, z1b, z2b, z3b,
                                                      c1, c2, c3,
                                                      rec2, rec3, r2, r3, Wb0,
                                                      cnt, gen, lossAcc, fllAcc);

  Ptrs p;
  p.x = x; p.Vw1 = Vw1; p.Vb1 = Vb1; p.Vb2 = Vb2; p.Vb3 = Vb3;
  p.Ww0 = Ww0; p.Wb0 = Wb0; p.Wb1 = Wb1; p.Wb2 = Wb2; p.r1 = r1;
  p.G1 = G1p; p.G2 = G2p; p.G3 = G3p; p.W1 = W1p; p.W2 = W2p; p.V2 = V2p; p.V3 = V3p;
  p.z1 = z1; p.z2 = z2; p.z3 = z3; p.g = g;
  p.c1 = c1; p.c2 = c2; p.c3 = c3;
  p.rec1 = rec1; p.rec2 = rec2; p.rec3 = rec3; p.td0 = td0;
  p.b1 = b1; p.b2 = b2; p.b3 = b3;
  p.z1b = z1b; p.z2b = z2b; p.z3b = z3b;
  p.cnt = cnt; p.gen = gen; p.lossAcc = lossAcc; p.fllAcc = fllAcc;
  p.out = (float*)d_out;

  predcells_persistent<<<NB, NT, 0, stream>>>(p);
}
